// Traveltime_32427003085121
// MI455X (gfx1250) — matrix operand order repaired and checked by reading, not compiled, not run on hardware
//
#include <hip/hip_runtime.h>
#include <hip/hip_bf16.h>

// Traveltime loss: per-trace max-abs normalize -> full xcorr (4095 lags) via
// bf16 WMMA (f32 accumulate) -> online softargmax -> (softargmax-2047)^2 -> sum.
//
// Tiling: D[M,N] = sum_k xp[ax0+32c+16M+k]*yp[ay0+32c+N+k] = cc[L0+16M-N],
// L0 = ax0-ay0.  16 lag blocks x 256 lags, one wave per block, per-block
// j-window trimming (576 K=32 chunks per trace instead of 1152).
// y is replicated 8x in LDS with +0..+7 element shifts so that every WMMA
// operand fetch is a 16B-aligned ds_load_b128.

#define NBATCH 8
#define NT     2048
#define NR     128
#define NC     16
#define NGROUP (NBATCH * NR * NC)   // 16384 traces
#define TRSTR  (NR * NC)            // 2048: time stride in elements

#define XOFF 240
#define XLEN 2528                   // logical x indices [-240, 2288)
#define YOFF 16
#define YLEN 2080                   // logical y indices [-16+s, ...) per shifted copy
#define BLOCKT 512                  // 16 waves of 32

typedef __attribute__((ext_vector_type(16))) __bf16 v16bf;
typedef __attribute__((ext_vector_type(8)))  __bf16 v8bf;
typedef __attribute__((ext_vector_type(8)))  float  v8f;

__global__ __launch_bounds__(BLOCKT)
void xcorr_softargmax_kernel(const float* __restrict__ x,
                             const float* __restrict__ y,
                             float* __restrict__ per_trace,   // NGROUP floats, or null
                             float* __restrict__ loss_atomic) // d_out, or null
{
    __shared__ __align__(16) __bf16 xbuf[XLEN];
    __shared__ __align__(16) __bf16 ybuf[8 * YLEN];
    __shared__ float stage[NT];       // raw f32 trace staging; reused as S/T scratch
    __shared__ float red[BLOCKT];     // max / m scratch

    const int tid = threadIdx.x;
    const int g   = blockIdx.x;
    const size_t gbase = (size_t)(g / TRSTR) * (size_t)NT * TRSTR + (size_t)(g % TRSTR);

    // ---------------- phase 0a: stage x, max-abs, normalize -> bf16 padded ----
    float lmax = 0.f;
    for (int t = tid; t < NT; t += BLOCKT) {
        float v = x[gbase + (size_t)t * TRSTR];
        stage[t] = v;
        lmax = fmaxf(lmax, fabsf(v));
    }
    red[tid] = lmax;
    __syncthreads();
    for (int s2 = BLOCKT / 2; s2 > 0; s2 >>= 1) {
        if (tid < s2) red[tid] = fmaxf(red[tid], red[tid + s2]);
        __syncthreads();
    }
    float invx = 1.0f / red[0];
    __syncthreads();
    for (int u = tid; u < XLEN; u += BLOCKT) {
        int j = u - XOFF;
        float v = (j >= 0 && j < NT) ? stage[j] * invx : 0.f;
        xbuf[u] = (__bf16)v;
    }
    __syncthreads();

    // ---------------- phase 0b: stage y, max-abs, 8 shifted bf16 copies ------
    lmax = 0.f;
    for (int t = tid; t < NT; t += BLOCKT) {
        float v = y[gbase + (size_t)t * TRSTR];
        stage[t] = v;
        lmax = fmaxf(lmax, fabsf(v));
    }
    red[tid] = lmax;
    __syncthreads();
    for (int s2 = BLOCKT / 2; s2 > 0; s2 >>= 1) {
        if (tid < s2) red[tid] = fmaxf(red[tid], red[tid + s2]);
        __syncthreads();
    }
    float invy = 1.0f / red[0];
    __syncthreads();
    for (int u = tid; u < 8 * YLEN; u += BLOCKT) {
        int s  = u / YLEN;
        int uu = u - s * YLEN;
        int j  = uu - YOFF + s;     // ybuf[s][uu] holds yp[uu - YOFF + s]
        float v = (j >= 0 && j < NT) ? stage[j] * invy : 0.f;
        ybuf[u] = (__bf16)v;
    }
    __syncthreads();

    // ---------------- phase 1: WMMA cross-correlation, one lag block / wave ---
    const int wave = tid >> 5;          // 0..15 == lag block b
    const int lane = tid & 31;
    const int mrow = lane & 15;         // M for A rows, N for B columns / D
    const int half = lane >> 4;         // K-half selector

    const int b   = wave;
    const int L0  = 256 * b - 2032;                              // block covers [L0-15, L0+240]
    const int ay0 = (b <= 7) ? (1792 - 256 * b) : -16;           // j-window base (mult of 16)
    const int ax0 = (b <= 7) ? -240 : (256 * b - 2048);          // = L0 + ay0  (mult of 16)
    const int Cn  = (b <= 7) ? (8 * (b + 1)) : (8 * (16 - b));   // K=32 chunk count

    const int s = mrow & 7;                                      // shifted y copy
    const __bf16* ybase = &ybuf[s * YLEN];

    v8f acc = {0.f, 0.f, 0.f, 0.f, 0.f, 0.f, 0.f, 0.f};

    int aoff = XOFF + ax0 + 16 * mrow + 16 * half;               // = 0 mod 8 elements
    int boff = ay0 + mrow + 16 * half + YOFF - s;                // = 0 mod 8 elements
    for (int c = 0; c < Cn; ++c) {
        v8bf a0 = *(const v8bf*)(xbuf + aoff);        // A: K = 16h+0..7
        v8bf a1 = *(const v8bf*)(xbuf + aoff + 8);    // A: K = 16h+8..15
        v8bf b0 = *(const v8bf*)(ybase + boff);       // B: K = 16h+0..7
        v8bf b1 = *(const v8bf*)(ybase + boff + 8);   // B: K = 16h+8..15
        v16bf Am = __builtin_shufflevector(a0, a1, 0,1,2,3,4,5,6,7,8,9,10,11,12,13,14,15);
        v16bf Bm = __builtin_shufflevector(b0, b1, 0,1,2,3,4,5,6,7,8,9,10,11,12,13,14,15);
        acc = __builtin_amdgcn_wmma_f32_16x16x32_bf16(false, Am, false, Bm,
                                                      (short)0, acc, false, false);
        aoff += 32;
        boff += 32;
    }

    // ---------------- phase 2: online softargmax over this thread's 8 lags ---
    float m = -3.4e38f, S = 0.f, T = 0.f;
    for (int r = 0; r < 8; ++r) {
        int l = L0 + 16 * (r + 8 * half) - mrow;     // D element r -> lag l
        if (l >= -(NT - 1) && l <= (NT - 1)) {       // drop the spurious l=2048 slot
            float ccv = acc[r];
            float kf  = (float)(l + (NT - 1));       // reference index 0..4094
            if (ccv > m) {
                float sc = __expf(m - ccv);
                S *= sc; T *= sc; m = ccv;
            }
            float e = __expf(ccv - m);
            S += e;
            T += e * kf;
        }
    }

    // workgroup-wide (m,S,T) merge: m in red[], S in stage[0..511], T in stage[512..]
    red[tid] = m;
    stage[tid] = S;
    stage[BLOCKT + tid] = T;
    __syncthreads();
    for (int s2 = BLOCKT / 2; s2 > 0; s2 >>= 1) {
        if (tid < s2) {
            float m1 = red[tid],        m2 = red[tid + s2];
            float S1 = stage[tid],      S2 = stage[tid + s2];
            float T1 = stage[BLOCKT + tid], T2 = stage[BLOCKT + tid + s2];
            float mm = fmaxf(m1, m2);
            float e1 = __expf(m1 - mm), e2 = __expf(m2 - mm);
            red[tid] = mm;
            stage[tid] = S1 * e1 + S2 * e2;
            stage[BLOCKT + tid] = T1 * e1 + T2 * e2;
        }
        __syncthreads();
    }
    if (tid == 0) {
        float sam = stage[BLOCKT] / stage[0];        // softargmax index in [0,4094]
        float d   = sam - (float)(NT - 1);
        float v   = d * d;
        if (per_trace) per_trace[blockIdx.x] = v;
        else           atomicAdd(loss_atomic, v);
    }
}

// Deterministic fixed-order final sum of per-trace losses.
__global__ __launch_bounds__(256)
void reduce_kernel(const float* __restrict__ per_trace, float* __restrict__ out)
{
    __shared__ float buf[256];
    int tid = threadIdx.x;
    float s = 0.f;
    for (int i = tid; i < NGROUP; i += 256) s += per_trace[i];
    buf[tid] = s;
    __syncthreads();
    for (int k = 128; k > 0; k >>= 1) {
        if (tid < k) buf[tid] += buf[tid + k];
        __syncthreads();
    }
    if (tid == 0) out[0] = buf[0];
}

__global__ void zero_kernel(float* out) { out[0] = 0.f; }

extern "C" void kernel_launch(void* const* d_in, const int* in_sizes, int n_in,
                              void* d_out, int out_size, void* d_ws, size_t ws_size,
                              hipStream_t stream)
{
    (void)in_sizes; (void)n_in; (void)out_size;
    const float* x = (const float*)d_in[0];
    const float* y = (const float*)d_in[1];
    float* out = (float*)d_out;

    if (d_ws && ws_size >= (size_t)NGROUP * sizeof(float)) {
        float* per_trace = (float*)d_ws;
        xcorr_softargmax_kernel<<<NGROUP, BLOCKT, 0, stream>>>(x, y, per_trace, nullptr);
        reduce_kernel<<<1, 256, 0, stream>>>(per_trace, out);
    } else {
        zero_kernel<<<1, 1, 0, stream>>>(out);
        xcorr_softargmax_kernel<<<NGROUP, BLOCKT, 0, stream>>>(x, y, nullptr, out);
    }
}
